// SelfAttention_17729624998294
// MI455X (gfx1250) — compile-verified, matrix operand order checked by reading, not run on hardware
//
#include <hip/hip_runtime.h>
#include <hip/hip_bf16.h>

// ---------------------------------------------------------------------------
// Self-attention (B=4, S=2048, D=1024, fp32 reference) for MI455X / gfx1250.
// Compute-bound (~120 GFLOP over ~80MB): v_wmma_f32_16x16x32_bf16 everywhere
// (bf16 inputs, f32 accumulate), wave32, 16x16x32 tiles.
//  - qkv_gemm: 4(M)x2(N) register blocking -> 8 WMMAs per 6 fragment loads.
//  - attention: flash-attention-2, 32-row q-blocks; every K/V B-fragment is
//    shared across two Q/P A-fragments, halving K/V L2 traffic vs 16-row
//    blocks. Un-normalized P staged in 32KB LDS; online softmax in fp32.
//  - bf16 conversion via native __bf16 casts (v_cvt_pk_bf16_f32, RNE).
// ---------------------------------------------------------------------------

#define D_DIM 1024
#define S_DIM 2048
#define B_DIM 4
#define BS    (B_DIM * S_DIM)          // 8192 rows
#define SCALE 0.03125f                 // 1/sqrt(1024)
#define CHUNK 256                      // keys per flash chunk
#define NCHUNK (S_DIM / CHUNK)         // 8

typedef __bf16 bf16x16 __attribute__((ext_vector_type(16)));
typedef unsigned short u16x16 __attribute__((ext_vector_type(16)));
typedef float f32x8 __attribute__((ext_vector_type(8)));

__device__ __forceinline__ unsigned short f2bf(float f) {
    __bf16 h = (__bf16)f;              // native cvt, round-to-nearest-even
    return __builtin_bit_cast(unsigned short, h);
}

__device__ __forceinline__ f32x8 zero8() {
    f32x8 z;
#pragma unroll
    for (int i = 0; i < 8; ++i) z[i] = 0.0f;
    return z;
}

__device__ __forceinline__ bf16x16 load_frag(const unsigned short* p) {
    u16x16 u = *(const u16x16*)p;      // 32B contiguous per lane
    return __builtin_bit_cast(bf16x16, u);
}

__device__ __forceinline__ f32x8 wmma_bf16(bf16x16 a, bf16x16 b, f32x8 c) {
    // (neg_a, A, neg_b, B, c_mod, C, reuse_a, reuse_b)
    return __builtin_amdgcn_wmma_f32_16x16x32_bf16(false, a, false, b,
                                                   (short)0, c, false, false);
}

// ---------------------------------------------------------------- converts --
__global__ void cvt_x_bf16(const float* __restrict__ in,
                           unsigned short* __restrict__ out, int n) {
    int i = blockIdx.x * blockDim.x + threadIdx.x;
    if (i < n) out[i] = f2bf(in[i]);
}

// W is [in=D][out=D]; store transposed bf16: wt[e][d] = W[d][e]
__global__ void cvt_w_transpose(const float* __restrict__ in,
                                unsigned short* __restrict__ out) {
    int idx = blockIdx.x * blockDim.x + threadIdx.x;   // 0 .. D*D-1
    int e = idx >> 10;
    int d = idx & (D_DIM - 1);
    out[idx] = f2bf(in[d * D_DIM + e]);
}

// ---------------------------------------------------------------- QKV GEMM --
// One wave computes a 64x32 output block as 4(M)x2(N) 16x16 WMMA tiles:
// per k-step, 4 A + 2 B fragment loads feed 8 WMMAs (0.75 loads/WMMA).
// sel=0 -> Q row-major, sel=1 -> K row-major, sel=2 -> V transposed [B][D][S].
__global__ __launch_bounds__(32)
void qkv_gemm(const unsigned short* __restrict__ xb,
              const unsigned short* __restrict__ wqt,
              const unsigned short* __restrict__ wkt,
              const unsigned short* __restrict__ wvt,
              unsigned short* __restrict__ qb,
              unsigned short* __restrict__ kb,
              unsigned short* __restrict__ vt) {
    const int lane = threadIdx.x;
    const int mt0  = blockIdx.x * 4;       // first of 4 M tiles (0..508)
    const int nt0  = blockIdx.y * 2;       // first of 2 N tiles (0..62)
    const int sel  = blockIdx.z;           // 0..2
    const unsigned short* wt = (sel == 0) ? wqt : ((sel == 1) ? wkt : wvt);

    const int l16  = lane & 15;
    const int lhi  = lane >> 4;
    const int koff = lhi * 16;

    size_t arow[4], brow[2];
#pragma unroll
    for (int i = 0; i < 4; ++i) arow[i] = (size_t)((mt0 + i) * 16 + l16) * D_DIM;
#pragma unroll
    for (int j = 0; j < 2; ++j) brow[j] = (size_t)((nt0 + j) * 16 + l16) * D_DIM;

    f32x8 acc[4][2];
#pragma unroll
    for (int i = 0; i < 4; ++i)
#pragma unroll
        for (int j = 0; j < 2; ++j) acc[i][j] = zero8();

    for (int k = 0; k < D_DIM; k += 32) {
        __builtin_prefetch(xb + arow[0] + k + 256, 0, 1);   // global_prefetch_b8
        bf16x16 a[4], b[2];
#pragma unroll
        for (int i = 0; i < 4; ++i) a[i] = load_frag(xb + arow[i] + k + koff);
#pragma unroll
        for (int j = 0; j < 2; ++j) b[j] = load_frag(wt + brow[j] + k + koff);
#pragma unroll
        for (int i = 0; i < 4; ++i)
#pragma unroll
            for (int j = 0; j < 2; ++j)
                acc[i][j] = wmma_bf16(a[i], b[j], acc[i][j]);
    }

#pragma unroll
    for (int i = 0; i < 4; ++i) {
        const int mbase = (mt0 + i) * 16 + 8 * lhi;
#pragma unroll
        for (int j = 0; j < 2; ++j) {
            const int n = (nt0 + j) * 16 + l16;
            if (sel == 2) {
#pragma unroll
                for (int r = 0; r < 8; ++r) {
                    int m  = mbase + r;            // global row in [0,8192)
                    int bb = m >> 11;              // batch
                    int s  = m & (S_DIM - 1);
                    vt[((size_t)bb * D_DIM + n) * S_DIM + s] = f2bf(acc[i][j][r]);
                }
            } else {
                unsigned short* o = (sel == 0) ? qb : kb;
#pragma unroll
                for (int r = 0; r < 8; ++r)
                    o[(size_t)(mbase + r) * D_DIM + n] = f2bf(acc[i][j][r]);
            }
        }
    }
}

// --------------------------------------------------------------- attention --
// 256-thread block (8 waves) owns 32 query rows (two 16-row halves) of one
// batch. Keys in NCHUNK chunks of CHUNK with online (flash) softmax.
// Scores: wave w computes keys [w*32, w*32+32) of the chunk for BOTH q-halves
//   (each K B-fragment feeds 2 WMMAs).
// P@V:    wave w computes output columns [w*128, w*128+128) for BOTH q-halves
//   (each V B-fragment feeds 2 WMMAs).
__global__ __launch_bounds__(256)
void attention(const unsigned short* __restrict__ qb,
               const unsigned short* __restrict__ kb,
               const unsigned short* __restrict__ vt,
               const int* __restrict__ mask,
               float* __restrict__ out) {
    __shared__ float sc[32 * CHUNK];      // 32 KB un-normalized scores / probs
    __shared__ float red[32 * 8];         // row reductions
    __shared__ float row_max[32], row_sum[32], fcorr[32], nmax[32];

    const int tid  = threadIdx.x;
    const int lane = tid & 31;
    const int wave = tid >> 5;            // 0..7
    const int qt   = blockIdx.x;          // 0..63
    const int bidx = blockIdx.y;          // 0..3
    const int q0   = qt * 32;
    const size_t rowbase = (size_t)bidx * S_DIM;

    if (tid < 32) { row_max[tid] = -1e30f; row_sum[tid] = 0.0f; }
    __syncthreads();

    const int l16  = lane & 15;
    const int lhi  = lane >> 4;           // 0/1
    const int koff = lhi * 16;

    f32x8 accO[2][8];
#pragma unroll
    for (int qh = 0; qh < 2; ++qh)
#pragma unroll
        for (int t = 0; t < 8; ++t) accO[qh][t] = zero8();

    const int r_ = tid >> 3;              // softmax row 0..31
    const int j_ = tid & 7;               // softmax col group
    const int CPT = CHUNK / 8;            // 32 cols per thread

    const size_t qrow0 = (rowbase + q0 + l16) * D_DIM;
    const size_t qrow1 = (rowbase + q0 + 16 + l16) * D_DIM;

    for (int c = 0; c < NCHUNK; ++c) {
        const int key0 = c * CHUNK;

        // ---- scores: S_tile = Q(32xD) * K^T, bf16 WMMA, f32 acc ----
        f32x8 accS[2][2];
#pragma unroll
        for (int qh = 0; qh < 2; ++qh)
#pragma unroll
            for (int t = 0; t < 2; ++t) accS[qh][t] = zero8();

        for (int k = 0; k < D_DIM; k += 32) {
            bf16x16 a0 = load_frag(qb + qrow0 + k + koff);
            bf16x16 a1 = load_frag(qb + qrow1 + k + koff);
#pragma unroll
            for (int t = 0; t < 2; ++t) {
                int key = key0 + wave * 32 + t * 16 + l16;
                bf16x16 b = load_frag(kb + (rowbase + key) * D_DIM + k + koff);
                accS[0][t] = wmma_bf16(a0, b, accS[0][t]);
                accS[1][t] = wmma_bf16(a1, b, accS[1][t]);
            }
        }
        // scale + mask bias, stage to LDS
#pragma unroll
        for (int t = 0; t < 2; ++t) {
            int kl  = wave * 32 + t * 16 + l16;        // col within chunk
            int key = key0 + kl;
            float bias = (mask[bidx * S_DIM + key] == 0) ? -1e9f : 0.0f;
#pragma unroll
            for (int qh = 0; qh < 2; ++qh)
#pragma unroll
                for (int r = 0; r < 8; ++r) {
                    int m = qh * 16 + r + 8 * lhi;
                    sc[m * CHUNK + kl] = accS[qh][t][r] * SCALE + bias;
                }
        }
        __syncthreads();

        // ---- online softmax over this chunk (32 rows x CHUNK) ----
        float lm = -1e30f;
        for (int cc = 0; cc < CPT; ++cc)
            lm = fmaxf(lm, sc[r_ * CHUNK + j_ * CPT + cc]);
        red[r_ * 8 + j_] = lm;
        __syncthreads();
        if (tid < 32) {
            float cm = -1e30f;
            for (int j = 0; j < 8; ++j) cm = fmaxf(cm, red[tid * 8 + j]);
            float om = row_max[tid];
            float nm = fmaxf(om, cm);
            nmax[tid]  = nm;
            fcorr[tid] = __expf(om - nm);
            row_max[tid] = nm;
        }
        __syncthreads();
        float nm = nmax[r_];
        float ls = 0.0f;
        for (int cc = 0; cc < CPT; ++cc) {
            int idx = r_ * CHUNK + j_ * CPT + cc;
            float e = __expf(sc[idx] - nm);
            sc[idx] = e;                  // in-place: un-normalized P
            ls += e;
        }
        red[r_ * 8 + j_] = ls;
        __syncthreads();
        if (tid < 32) {
            float csum = 0.0f;
            for (int j = 0; j < 8; ++j) csum += red[tid * 8 + j];
            row_sum[tid] = row_sum[tid] * fcorr[tid] + csum;
        }
        __syncthreads();

        // ---- rescale running output accumulators by exp(old_max-new_max) ----
        float fr[2][8];
#pragma unroll
        for (int qh = 0; qh < 2; ++qh)
#pragma unroll
            for (int r = 0; r < 8; ++r) fr[qh][r] = fcorr[qh * 16 + r + 8 * lhi];
#pragma unroll
        for (int qh = 0; qh < 2; ++qh)
#pragma unroll
            for (int t = 0; t < 8; ++t)
#pragma unroll
                for (int r = 0; r < 8; ++r) accO[qh][t][r] *= fr[qh][r];

        // ---- accumulate O += P(32xCHUNK) * V(CHUNKx128) for this wave ----
        for (int sb = 0; sb < CHUNK; sb += 32) {
            // A fragments from LDS (fp32 -> bf16 via v_cvt_pk_bf16_f32)
            bf16x16 a2[2];
#pragma unroll
            for (int qh = 0; qh < 2; ++qh)
#pragma unroll
                for (int i = 0; i < 16; ++i)
                    a2[qh][i] =
                        (__bf16)sc[(qh * 16 + l16) * CHUNK + sb + koff + i];
            int s = key0 + sb + koff;
#pragma unroll
            for (int t = 0; t < 8; ++t) {
                int n = wave * 128 + t * 16 + l16;
                bf16x16 b =
                    load_frag(vt + ((size_t)bidx * D_DIM + n) * S_DIM + s);
                accO[0][t] = wmma_bf16(a2[0], b, accO[0][t]);
                accO[1][t] = wmma_bf16(a2[1], b, accO[1][t]);
            }
        }
        __syncthreads();   // all waves done with sc before next chunk
    }

    // ---- normalize and write output (fp32) ----
    float rs[2][8];
#pragma unroll
    for (int qh = 0; qh < 2; ++qh)
#pragma unroll
        for (int r = 0; r < 8; ++r)
            rs[qh][r] = 1.0f / row_sum[qh * 16 + r + 8 * lhi];
#pragma unroll
    for (int qh = 0; qh < 2; ++qh)
#pragma unroll
        for (int t = 0; t < 8; ++t) {
            int n = wave * 128 + t * 16 + l16;
#pragma unroll
            for (int r = 0; r < 8; ++r) {
                int m = q0 + qh * 16 + r + 8 * lhi;
                out[(rowbase + m) * D_DIM + n] = accO[qh][t][r] * rs[qh][r];
            }
        }
}

// ------------------------------------------------------------------ launch --
extern "C" void kernel_launch(void* const* d_in, const int* in_sizes, int n_in,
                              void* d_out, int out_size, void* d_ws, size_t ws_size,
                              hipStream_t stream) {
    const float* x    = (const float*)d_in[0];
    const int*   mask = (const int*)d_in[1];
    const float* Wq   = (const float*)d_in[2];
    const float* Wk   = (const float*)d_in[3];
    const float* Wv   = (const float*)d_in[4];
    float* out = (float*)d_out;

    // workspace layout (bf16 = ushort), ~73.4 MB total
    char* ws = (char*)d_ws;
    const size_t XB   = (size_t)BS * D_DIM * 2;       // 16 MB
    const size_t WT   = (size_t)D_DIM * D_DIM * 2;    // 2 MB
    unsigned short* xb  = (unsigned short*)(ws);
    unsigned short* wqt = (unsigned short*)(ws + XB);
    unsigned short* wkt = (unsigned short*)(ws + XB + WT);
    unsigned short* wvt = (unsigned short*)(ws + XB + 2 * WT);
    unsigned short* qb  = (unsigned short*)(ws + XB + 3 * WT);
    unsigned short* kb  = (unsigned short*)(ws + 2 * XB + 3 * WT);
    unsigned short* vt  = (unsigned short*)(ws + 3 * XB + 3 * WT);

    const int nx = BS * D_DIM;                         // 8,388,608
    cvt_x_bf16<<<nx / 256, 256, 0, stream>>>(x, xb, nx);
    const int nw = D_DIM * D_DIM;                      // 1,048,576
    cvt_w_transpose<<<nw / 256, 256, 0, stream>>>(Wq, wqt);
    cvt_w_transpose<<<nw / 256, 256, 0, stream>>>(Wk, wkt);
    cvt_w_transpose<<<nw / 256, 256, 0, stream>>>(Wv, wvt);

    dim3 ggrid(BS / 64, D_DIM / 32, 3);                // 128 x 32 x 3
    qkv_gemm<<<ggrid, 32, 0, stream>>>(xb, wqt, wkt, wvt, qb, kb, vt);

    dim3 agrid(S_DIM / 32, B_DIM);                     // 64 x 4
    attention<<<agrid, 256, 0, stream>>>(qb, kb, vt, mask, out);
}
